// NN_Model_11897059410605
// MI455X (gfx1250) — compile-verified, matrix operand order checked by reading, not run on hardware
//
#include <hip/hip_runtime.h>

typedef unsigned int   u32;
typedef unsigned short u16;

typedef __bf16 bf16;
typedef bf16  v16bf __attribute__((ext_vector_type(16)));
typedef float v8f   __attribute__((ext_vector_type(8)));

// ---------- constants (from reference) ----------
constexpr int N_MOL  = 24 * 48;    // 1152
constexpr int N_PRO  = 24 * 256;   // 6144
constexpr int N_ALL  = N_MOL + N_PRO; // 7296
constexpr int HD     = 128;
constexpr int KIN_E  = 288;        // 272 padded to 9*32
constexpr int KIN_N  = 256;

// workspace layout (bytes)
constexpr size_t OFF_H    = 0;
constexpr size_t OFF_AGG  = (size_t)N_ALL * HD * 4;                  // 3,735,552
constexpr size_t OFF_HBF  = OFF_AGG + (size_t)N_ALL * HD * 4;        // 7,471,104
constexpr size_t OFF_FRAG = OFF_HBF + (size_t)N_ALL * HD * 2;        // 9,338,880
// per-layer fragment region (u16 units): ew1 | ew2 | nw1 | nw2
constexpr int FRAG_LAYER_STRIDE = 102400;  // u16
constexpr int FRAG_EW2_OFF = 36864;        // u16
constexpr int FRAG_NW1_OFF = 53248;        // u16
constexpr int FRAG_NW2_OFF = 86016;        // u16

// ---------- helpers ----------
struct __align__(16) Q16 { u32 u[4]; };
struct Q32 { Q16 a, b; };

static __device__ inline v16bf frag_cast(Q16 a, Q16 b) {
    Q32 q{a, b};
    return __builtin_bit_cast(v16bf, q);
}

static __device__ inline u16 f2bf(float f) {
    u32 u = __builtin_bit_cast(u32, f);
    u += 0x7FFFu + ((u >> 16) & 1u);           // round-to-nearest-even
    return (u16)(u >> 16);
}

static __device__ inline float silu(float x) { return x / (1.0f + __expf(-x)); }

static __device__ inline v8f wmma_bf16(v16bf a, v16bf b, v8f c) {
    // v_wmma_f32_16x16x32_bf16: D = A(16x32) * B(32x16) + C(16x16 f32)
    return __builtin_amdgcn_wmma_f32_16x16x32_bf16(false, a, false, b, (short)0, c,
                                                   false, false);
}

static __device__ inline v16bf load_b_frag(const u16* fb, int kt, int nt, int lane) {
    const Q16* p = reinterpret_cast<const Q16*>(fb + (((kt * 8 + nt) * 32 + lane) << 4));
    return frag_cast(p[0], p[1]);
}

// ---------- weight fragment pre-pack ----------
// B-operand layout for 16-bit WMMA (ISA 7.12.2 mirror of A layout):
//   lane<16 : N = nt*16+lane,    elems 0..7 -> K=kt*32+0..7,  8..15 -> K+16..23
//   lane>=16: N = nt*16+lane-16, elems 0..7 -> K=kt*32+8..15, 8..15 -> K+24..31
__global__ __launch_bounds__(256) void prep_kernel(
    const float* __restrict__ ew1, const float* __restrict__ ew2,
    const float* __restrict__ nw1, const float* __restrict__ nw2,
    u16* __restrict__ frag)
{
    int tid = blockIdx.x * 256 + threadIdx.x;
    if (tid >= 4 * 200 * 32) return;
    int lane  = tid & 31;
    int f     = (tid >> 5) % 200;
    int layer = tid / (200 * 32);
    u16* L = frag + (size_t)layer * FRAG_LAYER_STRIDE;
    const float* W; int Ksrc, fr; u16* dst;
    if (f < 72)       { W = ew1 + (size_t)layer * 272 * 128; Ksrc = 272; fr = f;       dst = L; }
    else if (f < 104) { W = ew2 + (size_t)layer * 128 * 128; Ksrc = 128; fr = f - 72;  dst = L + FRAG_EW2_OFF; }
    else if (f < 168) { W = nw1 + (size_t)layer * 256 * 128; Ksrc = 256; fr = f - 104; dst = L + FRAG_NW1_OFF; }
    else              { W = nw2 + (size_t)layer * 128 * 128; Ksrc = 128; fr = f - 168; dst = L + FRAG_NW2_OFF; }
    int kt = fr >> 3, nt = fr & 7;
    int nn = nt * 16 + (lane & 15);
    u16* d = dst + (((size_t)fr * 32 + lane) << 4);
    #pragma unroll
    for (int e = 0; e < 16; ++e) {
        int kl = e + ((e >= 8) ? 8 : 0) + ((lane >= 16) ? 8 : 0);
        int k = kt * 32 + kl;
        float v = (k < Ksrc) ? W[k * 128 + nn] : 0.0f;
        d[e] = f2bf(v);
    }
}

// ---------- node embedding: h0 = [x, MLP(feat), t] @ gin_w + gin_b ----------
__global__ __launch_bounds__(256) void embed_kernel(
    const float* __restrict__ zm, const float* __restrict__ zp,
    const float* __restrict__ t,
    const float* __restrict__ aw1, const float* __restrict__ ab1,
    const float* __restrict__ aw2, const float* __restrict__ ab2,
    const float* __restrict__ rw1, const float* __restrict__ rb1,
    const float* __restrict__ rw2, const float* __restrict__ rb2,
    const float* __restrict__ ginw, const float* __restrict__ ginb,
    float* __restrict__ h, u16* __restrict__ hbf, float* __restrict__ agg)
{
    int n = blockIdx.x * 256 + threadIdx.x;
    if (n >= N_ALL) return;
    const float *z, *w1, *b1, *w2, *b2;
    float tv;
    if (n < N_MOL) { z = zm + (size_t)n * 23; w1 = aw1; b1 = ab1; w2 = aw2; b2 = ab2; tv = t[n / 48]; }
    else { int p = n - N_MOL; z = zp + (size_t)p * 23; w1 = rw1; b1 = rb1; w2 = rw2; b2 = rb2; tv = t[p / 256]; }

    float in36[36];
    in36[0] = z[0]; in36[1] = z[1]; in36[2] = z[2];
    float hd40[40];
    #pragma unroll
    for (int j = 0; j < 40; ++j) {
        float s = b1[j];
        #pragma unroll
        for (int i = 0; i < 20; ++i) s += z[3 + i] * w1[i * 40 + j];
        hd40[j] = silu(s);
    }
    #pragma unroll
    for (int j = 0; j < 32; ++j) {
        float s = b2[j];
        #pragma unroll
        for (int i = 0; i < 40; ++i) s += hd40[i] * w2[i * 32 + j];
        in36[3 + j] = s;
    }
    in36[35] = tv;
    for (int o = 0; o < 128; ++o) {
        float s = ginb[o];
        #pragma unroll
        for (int k = 0; k < 36; ++k) s += in36[k] * ginw[k * 128 + o];
        size_t idx = (size_t)n * 128 + o;
        h[idx] = s; hbf[idx] = f2bf(s); agg[idx] = 0.0f;
    }
}

// ---------- fused edge message + scatter-add ----------
// 2 waves/block, 32 edges/wave (2 M-tiles share B fragments).
// B fragments double-buffered across kt to overlap L2 latency with WMMAs.
__global__ __launch_bounds__(64) void edge_kernel(
    const u16* __restrict__ hbf, const int* __restrict__ edges, int E,
    const float* __restrict__ eemb,
    const u16* __restrict__ fW1, const u16* __restrict__ fW2,
    const float* __restrict__ b1, const float* __restrict__ b2,
    float* __restrict__ agg)
{
    __shared__ __align__(16) u16 sIn[2][32][KIN_E];
    __shared__ __align__(16) u16 sHid[2][32][HD];
    __shared__ int sRow[2][32];
    __shared__ int sCol[2][32];

    const int wid = threadIdx.x >> 5;
    const int lane = threadIdx.x & 31;
    const int tile = (blockIdx.x * 2 + wid) * 32;

    { // indices
        int e = tile + lane; int r = -1, c = -1;
        if (e < E) { r = edges[e]; c = edges[E + e]; }
        sRow[wid][lane] = r; sCol[wid][lane] = c;
    }
    // stage h[row] | h[col] (bf16, copied as dwords)
    const u32* hbfU = reinterpret_cast<const u32*>(hbf);
    for (int idx = lane; idx < 32 * 128; idx += 32) {
        int slot = idx >> 7, u = idx & 127;
        int nd = (u < 64) ? sRow[wid][slot] : sCol[wid][slot];
        u32 v = 0;
        if (nd >= 0) v = hbfU[(size_t)nd * 64 + (u & 63)];
        reinterpret_cast<u32*>(&sIn[wid][slot][0])[u] = v;
    }
    // edge_attr + zero pad (cols 256..287)
    for (int idx = lane; idx < 32 * 32; idx += 32) {
        int slot = idx >> 5, j = idx & 31;
        int r = sRow[wid][slot], c = sCol[wid][slot];
        u16 v = 0;
        if (r >= 0 && j < 16) {
            int et = (r < N_MOL && c < N_MOL) ? 1 : ((r >= N_MOL && c >= N_MOL) ? 2 : 0);
            v = f2bf(eemb[et * 16 + j]);
        }
        sIn[wid][slot][256 + j] = v;
    }

    const int l15 = lane & 15, hl = lane >> 4;
    v8f acc[2][8];
    #pragma unroll
    for (int m2 = 0; m2 < 2; ++m2)
        #pragma unroll
        for (int nt = 0; nt < 8; ++nt)
            #pragma unroll
            for (int r = 0; r < 8; ++r) acc[m2][nt][r] = 0.0f;

    v16bf bcur[8], bnxt[8];

    // GEMM1: in(32x288) @ ew1(288x128), B double-buffered over kt
    #pragma unroll
    for (int nt = 0; nt < 8; ++nt) bcur[nt] = load_b_frag(fW1, 0, nt, lane);
    #pragma unroll
    for (int kt = 0; kt < 9; ++kt) {
        int k0 = kt * 32 + (hl ? 8 : 0);
        v16bf a0 = frag_cast(*reinterpret_cast<const Q16*>(&sIn[wid][l15][k0]),
                             *reinterpret_cast<const Q16*>(&sIn[wid][l15][k0 + 16]));
        v16bf a1 = frag_cast(*reinterpret_cast<const Q16*>(&sIn[wid][16 + l15][k0]),
                             *reinterpret_cast<const Q16*>(&sIn[wid][16 + l15][k0 + 16]));
        if (kt < 8) {
            #pragma unroll
            for (int nt = 0; nt < 8; ++nt) bnxt[nt] = load_b_frag(fW1, kt + 1, nt, lane);
        }
        #pragma unroll
        for (int nt = 0; nt < 8; ++nt) {
            acc[0][nt] = wmma_bf16(a0, bcur[nt], acc[0][nt]);
            acc[1][nt] = wmma_bf16(a1, bcur[nt], acc[1][nt]);
        }
        if (kt < 8) {
            #pragma unroll
            for (int nt = 0; nt < 8; ++nt) bcur[nt] = bnxt[nt];
        }
    }
    // bias + silu -> sHid (bf16)
    #pragma unroll
    for (int m2 = 0; m2 < 2; ++m2)
        #pragma unroll
        for (int nt = 0; nt < 8; ++nt) {
            int n = nt * 16 + l15;
            float bb = b1[n];
            #pragma unroll
            for (int r = 0; r < 8; ++r)
                sHid[wid][m2 * 16 + r + 8 * hl][n] = f2bf(silu(acc[m2][nt][r] + bb));
        }
    #pragma unroll
    for (int m2 = 0; m2 < 2; ++m2)
        #pragma unroll
        for (int nt = 0; nt < 8; ++nt)
            #pragma unroll
            for (int r = 0; r < 8; ++r) acc[m2][nt][r] = 0.0f;

    // GEMM2: hid(32x128) @ ew2(128x128), B double-buffered over kt
    #pragma unroll
    for (int nt = 0; nt < 8; ++nt) bcur[nt] = load_b_frag(fW2, 0, nt, lane);
    #pragma unroll
    for (int kt = 0; kt < 4; ++kt) {
        int k0 = kt * 32 + (hl ? 8 : 0);
        v16bf a0 = frag_cast(*reinterpret_cast<const Q16*>(&sHid[wid][l15][k0]),
                             *reinterpret_cast<const Q16*>(&sHid[wid][l15][k0 + 16]));
        v16bf a1 = frag_cast(*reinterpret_cast<const Q16*>(&sHid[wid][16 + l15][k0]),
                             *reinterpret_cast<const Q16*>(&sHid[wid][16 + l15][k0 + 16]));
        if (kt < 3) {
            #pragma unroll
            for (int nt = 0; nt < 8; ++nt) bnxt[nt] = load_b_frag(fW2, kt + 1, nt, lane);
        }
        #pragma unroll
        for (int nt = 0; nt < 8; ++nt) {
            acc[0][nt] = wmma_bf16(a0, bcur[nt], acc[0][nt]);
            acc[1][nt] = wmma_bf16(a1, bcur[nt], acc[1][nt]);
        }
        if (kt < 3) {
            #pragma unroll
            for (int nt = 0; nt < 8; ++nt) bcur[nt] = bnxt[nt];
        }
    }

    // m = silu(. + b2)/NORM, scatter-add into agg[row] with run-merging
    #pragma unroll
    for (int m2 = 0; m2 < 2; ++m2)
        #pragma unroll
        for (int nt = 0; nt < 8; ++nt) {
            int n = nt * 16 + l15;
            float bb = b2[n];
            float pend = 0.0f; int prow = -1;
            #pragma unroll
            for (int r = 0; r < 8; ++r) {
                int slot = m2 * 16 + r + 8 * hl;
                int rw = sRow[wid][slot];
                if (rw != prow) {
                    if (prow >= 0) atomicAdd(&agg[(size_t)prow * 128 + n], pend);
                    prow = rw; pend = 0.0f;
                }
                if (rw >= 0) pend += silu(acc[m2][nt][r] + bb) * 0.01f;  // /NORM
            }
            if (prow >= 0) atomicAdd(&agg[(size_t)prow * 128 + n], pend);
        }
}

// ---------- fused node update: h += MLP([h, agg]); re-zero agg ----------
__global__ __launch_bounds__(128) void node_kernel(
    float* __restrict__ h, u16* __restrict__ hbf, float* __restrict__ agg,
    const u16* __restrict__ fW1, const u16* __restrict__ fW2,
    const float* __restrict__ b1, const float* __restrict__ b2)
{
    __shared__ __align__(16) u16 sIn[4][16][KIN_N];
    __shared__ __align__(16) u16 sHid[4][16][HD];

    const int wid = threadIdx.x >> 5;
    const int lane = threadIdx.x & 31;
    const int base = (blockIdx.x * 4 + wid) * 16;

    const u32* hbfU = reinterpret_cast<const u32*>(hbf);
    for (int idx = lane; idx < 16 * 64; idx += 32) {
        int slot = idx >> 6, u = idx & 63;
        reinterpret_cast<u32*>(&sIn[wid][slot][0])[u] = hbfU[(size_t)(base + slot) * 64 + u];
    }
    for (int idx = lane; idx < 16 * 128; idx += 32) {
        int slot = idx >> 7, j = idx & 127;
        sIn[wid][slot][128 + j] = f2bf(agg[(size_t)(base + slot) * 128 + j]);
    }

    const int l15 = lane & 15, hl = lane >> 4;
    v8f acc[8];
    #pragma unroll
    for (int nt = 0; nt < 8; ++nt)
        #pragma unroll
        for (int r = 0; r < 8; ++r) acc[nt][r] = 0.0f;

    v16bf bcur[8], bnxt[8];

    // GEMM1: [h,agg](16x256) @ nw1(256x128), B double-buffered over kt
    #pragma unroll
    for (int nt = 0; nt < 8; ++nt) bcur[nt] = load_b_frag(fW1, 0, nt, lane);
    #pragma unroll
    for (int kt = 0; kt < 8; ++kt) {
        int k0 = kt * 32 + (hl ? 8 : 0);
        v16bf a = frag_cast(*reinterpret_cast<const Q16*>(&sIn[wid][l15][k0]),
                            *reinterpret_cast<const Q16*>(&sIn[wid][l15][k0 + 16]));
        if (kt < 7) {
            #pragma unroll
            for (int nt = 0; nt < 8; ++nt) bnxt[nt] = load_b_frag(fW1, kt + 1, nt, lane);
        }
        #pragma unroll
        for (int nt = 0; nt < 8; ++nt) acc[nt] = wmma_bf16(a, bcur[nt], acc[nt]);
        if (kt < 7) {
            #pragma unroll
            for (int nt = 0; nt < 8; ++nt) bcur[nt] = bnxt[nt];
        }
    }
    #pragma unroll
    for (int nt = 0; nt < 8; ++nt) {
        int n = nt * 16 + l15;
        float bb = b1[n];
        #pragma unroll
        for (int r = 0; r < 8; ++r)
            sHid[wid][r + 8 * hl][n] = f2bf(silu(acc[nt][r] + bb));
    }
    #pragma unroll
    for (int nt = 0; nt < 8; ++nt)
        #pragma unroll
        for (int r = 0; r < 8; ++r) acc[nt][r] = 0.0f;

    // GEMM2: hid(16x128) @ nw2(128x128), B double-buffered over kt
    #pragma unroll
    for (int nt = 0; nt < 8; ++nt) bcur[nt] = load_b_frag(fW2, 0, nt, lane);
    #pragma unroll
    for (int kt = 0; kt < 4; ++kt) {
        int k0 = kt * 32 + (hl ? 8 : 0);
        v16bf a = frag_cast(*reinterpret_cast<const Q16*>(&sHid[wid][l15][k0]),
                            *reinterpret_cast<const Q16*>(&sHid[wid][l15][k0 + 16]));
        if (kt < 3) {
            #pragma unroll
            for (int nt = 0; nt < 8; ++nt) bnxt[nt] = load_b_frag(fW2, kt + 1, nt, lane);
        }
        #pragma unroll
        for (int nt = 0; nt < 8; ++nt) acc[nt] = wmma_bf16(a, bcur[nt], acc[nt]);
        if (kt < 3) {
            #pragma unroll
            for (int nt = 0; nt < 8; ++nt) bcur[nt] = bnxt[nt];
        }
    }
    // h += delta; refresh bf16 mirror; zero agg for next layer
    #pragma unroll
    for (int nt = 0; nt < 8; ++nt) {
        int n = nt * 16 + l15;
        float bb = b2[n];
        #pragma unroll
        for (int r = 0; r < 8; ++r) {
            size_t idx = (size_t)(base + r + 8 * hl) * 128 + n;
            float nh = h[idx] + acc[nt][r] + bb;
            h[idx] = nh; hbf[idx] = f2bf(nh); agg[idx] = 0.0f;
        }
    }
}

// ---------- output head ----------
__global__ __launch_bounds__(256) void out_kernel(
    const float* __restrict__ h, const float* __restrict__ gw, const float* __restrict__ gb,
    const float* __restrict__ adw1, const float* __restrict__ adb1,
    const float* __restrict__ adw2, const float* __restrict__ adb2,
    const float* __restrict__ rdw1, const float* __restrict__ rdb1,
    const float* __restrict__ rdw2, const float* __restrict__ rdb2,
    float* __restrict__ out)
{
    int n = blockIdx.x * 256 + threadIdx.x;
    if (n >= N_ALL) return;
    const float* hr = h + (size_t)n * 128;
    float o36[36];
    for (int o = 0; o < 36; ++o) {
        float s = gb[o];
        for (int k = 0; k < 128; ++k) s += hr[k] * gw[k * 36 + o];
        o36[o] = s;
    }
    const float *w1, *b1, *w2, *b2;
    if (n < N_MOL) { w1 = adw1; b1 = adb1; w2 = adw2; b2 = adb2; }
    else           { w1 = rdw1; b1 = rdb1; w2 = rdw2; b2 = rdb2; }
    float hd40[40];
    #pragma unroll
    for (int j = 0; j < 40; ++j) {
        float s = b1[j];
        #pragma unroll
        for (int i = 0; i < 32; ++i) s += o36[3 + i] * w1[i * 40 + j];
        hd40[j] = silu(s);
    }
    float* dst = out + (size_t)n * 23;   // eps_mol then eps_pro are contiguous rows of 23
    dst[0] = o36[0]; dst[1] = o36[1]; dst[2] = o36[2];
    #pragma unroll
    for (int j = 0; j < 20; ++j) {
        float s = b2[j];
        #pragma unroll
        for (int i = 0; i < 40; ++i) s += hd40[i] * w2[i * 20 + j];
        dst[3 + j] = s;
    }
}

// ---------- launcher ----------
extern "C" void kernel_launch(void* const* d_in, const int* in_sizes, int n_in,
                              void* d_out, int out_size, void* d_ws, size_t ws_size,
                              hipStream_t stream) {
    const float* zm   = (const float*)d_in[0];
    const float* zp   = (const float*)d_in[1];
    const float* t    = (const float*)d_in[2];
    const int*   edges= (const int*)  d_in[5];
    const float* aw1  = (const float*)d_in[6];
    const float* ab1  = (const float*)d_in[7];
    const float* aw2  = (const float*)d_in[8];
    const float* ab2  = (const float*)d_in[9];
    const float* rw1  = (const float*)d_in[10];
    const float* rb1  = (const float*)d_in[11];
    const float* rw2  = (const float*)d_in[12];
    const float* rb2  = (const float*)d_in[13];
    const float* adw1 = (const float*)d_in[14];
    const float* adb1 = (const float*)d_in[15];
    const float* adw2 = (const float*)d_in[16];
    const float* adb2 = (const float*)d_in[17];
    const float* rdw1 = (const float*)d_in[18];
    const float* rdb1 = (const float*)d_in[19];
    const float* rdw2 = (const float*)d_in[20];
    const float* rdb2 = (const float*)d_in[21];
    const float* eemb = (const float*)d_in[22];
    const float* ginw = (const float*)d_in[23];
    const float* ginb = (const float*)d_in[24];
    const float* goutw= (const float*)d_in[25];
    const float* goutb= (const float*)d_in[26];
    const float* ew1  = (const float*)d_in[27];
    const float* eb1  = (const float*)d_in[28];
    const float* ew2  = (const float*)d_in[29];
    const float* eb2  = (const float*)d_in[30];
    const float* nw1  = (const float*)d_in[31];
    const float* nb1  = (const float*)d_in[32];
    const float* nw2  = (const float*)d_in[33];
    const float* nb2  = (const float*)d_in[34];
    float* outp = (float*)d_out;

    const int E = in_sizes[5] / 2;
    char* ws = (char*)d_ws;
    float* h    = (float*)(ws + OFF_H);
    float* agg  = (float*)(ws + OFF_AGG);
    u16*   hbf  = (u16*)  (ws + OFF_HBF);
    u16*   frag = (u16*)  (ws + OFF_FRAG);

    prep_kernel<<<100, 256, 0, stream>>>(ew1, ew2, nw1, nw2, frag);
    embed_kernel<<<(N_ALL + 255) / 256, 256, 0, stream>>>(
        zm, zp, t, aw1, ab1, aw2, ab2, rw1, rb1, rw2, rb2, ginw, ginb, h, hbf, agg);

    for (int l = 0; l < 4; ++l) {
        u16* fL = frag + (size_t)l * FRAG_LAYER_STRIDE;
        if (E > 0) {
            edge_kernel<<<(E + 63) / 64, 64, 0, stream>>>(
                hbf, edges, E, eemb, fL, fL + FRAG_EW2_OFF,
                eb1 + l * 128, eb2 + l * 128, agg);
        }
        node_kernel<<<N_ALL / 64, 128, 0, stream>>>(
            h, hbf, agg, fL + FRAG_NW1_OFF, fL + FRAG_NW2_OFF,
            nb1 + l * 128, nb2 + l * 128);
    }

    out_kernel<<<(N_ALL + 255) / 256, 256, 0, stream>>>(
        h, goutw, goutb, adw1, adb1, adw2, adb2, rdw1, rdb1, rdw2, rdb2, outp);
}
